// Attention_33406255628438
// MI455X (gfx1250) — matrix operand order repaired and checked by reading, not compiled, not run on hardware
//
#include <hip/hip_runtime.h>
#include <hip/hip_bf16.h>

// ---------------------------------------------------------------------------
// Talking-heads attention, fused for MI455X (gfx1250, wave32, WMMA bf16).
//   K1: qkv projection GEMM  (f32 in -> bf16 q,k,vT in workspace)
//   K2: two-pass flash attention with pre/post head mixing (12 waves = heads)
//       + async (ASYNCcnt) LDS staging of V tiles overlapped with softmax
//   K3: output projection GEMM (+bias, f32 out)
// Workspace use: 4*(6 MB) = 24 MB  (q, k, vT, ctx), all bf16.
// ---------------------------------------------------------------------------

typedef __bf16 bf16;
typedef __attribute__((ext_vector_type(16))) __bf16 v16bf;
typedef __attribute__((ext_vector_type(8)))  __bf16 v8bf;
typedef __attribute__((ext_vector_type(4)))  __bf16 v4bf;
typedef __attribute__((ext_vector_type(2)))  __bf16 v2bf;
typedef __attribute__((ext_vector_type(8)))  float  v8f;
typedef __attribute__((ext_vector_type(4)))  float  f32x4;
typedef __attribute__((ext_vector_type(2)))  float  f32x2;

union V16U { v16bf v; v8bf h[2]; };

#define NHEADS 12
#define DHEAD  64
#define SEQ    1024
#define NBATCH 4
#define DIM    768
#define INNER  768
#define QKSCALE 0.125f /* 64^-0.5 */

__device__ __forceinline__ bf16 tobf(float f) { return (bf16)f; }

__device__ __forceinline__ v8f v8f_zero() {
  v8f r;
#pragma unroll
  for (int i = 0; i < 8; ++i) r[i] = 0.f;
  return r;
}

// A-operand (16x32 bf16) k order used here (contiguous, B uses the same): VGPR v holds K pair starting at:
//   lanes 0-15 : 2*v ; lanes 16-31: 16 + 2*v   (row = lane&15)
__device__ __forceinline__ int a_kbase(int v, int hi) {
  return 2 * v + (hi ? 16 : 0);
}

// wait for all outstanding async (ASYNCcnt) LDS copies
__device__ __forceinline__ void wait_async0() {
#if __has_builtin(__builtin_amdgcn_s_wait_asynccnt)
  __builtin_amdgcn_s_wait_asynccnt(0);
#else
  asm volatile("s_wait_asynccnt 0x0" ::: "memory");
#endif
}

// ---------------------------------------------------------------------------
// K1: [4096 x 768] @ [768 x 2304] -> q(scaled), k, v(transposed), bf16.
// Block: 256 thr (8 waves). Tile: BM=128 (16 rows/wave), BN=64, BK=32.
// A read direct from global (rows are wave-private), B staged in LDS
// transposed to [col][k] so B-fragments are contiguous 16-elem LDS reads.
// ---------------------------------------------------------------------------
__global__ __launch_bounds__(256) void qkv_gemm(
    const float* __restrict__ x, const float* __restrict__ Wq,
    const float* __restrict__ Wkv, bf16* __restrict__ q,
    bf16* __restrict__ k, bf16* __restrict__ vT) {
  __shared__ bf16 Bs[64 * 40];  // [col][kk], stride 40 (80B rows, bank-friendly)

  const int tid = threadIdx.x;
  const int lane = tid & 31, wave = tid >> 5;
  const int hi = (lane >> 4) & 1, ln = lane & 15;
  const int cb = blockIdx.x * 64;  // global output column base (0..2240)

  const float* W;
  int ldw, colBase;
  if (cb < 768) { W = Wq;  ldw = INNER;     colBase = cb; }
  else          { W = Wkv; ldw = 2 * INNER; colBase = cb - 768; }

  const int mrow = blockIdx.y * 128 + wave * 16 + ln;  // A-fragment row
  const int scol = tid & 63;                           // staging column
  const int soct = tid >> 6;                           // staging k-octet (0..3)

  v8f acc[4];
#pragma unroll
  for (int ng = 0; ng < 4; ++ng) acc[ng] = v8f_zero();

  for (int ks = 0; ks < DIM; ks += 32) {
    // ---- stage B tile: Bs[col][k] (each thread: 8 strided f32 -> 1 b128 ds)
    {
      bf16 tmp[8];
      const float* wp = &W[(size_t)(ks + soct * 8) * ldw + colBase + scol];
#pragma unroll
      for (int j = 0; j < 8; ++j) tmp[j] = tobf(wp[(size_t)j * ldw]);
      *(v8bf*)&Bs[scol * 40 + soct * 8] = *(const v8bf*)tmp;
      if (ks + 32 < DIM)
        __builtin_prefetch(&W[(size_t)(ks + 32 + soct * 8) * ldw + colBase + scol], 0, 1);
    }
    __syncthreads();

    // ---- A fragment straight from global x (f32 pairs -> bf16)
    v16bf a;
#pragma unroll
    for (int v = 0; v < 8; ++v) {
      f32x2 p = *(const f32x2*)&x[(size_t)mrow * DIM + ks + a_kbase(v, hi)];
      a[2 * v] = tobf(p[0]);
      a[2 * v + 1] = tobf(p[1]);
    }

#pragma unroll
    for (int ng = 0; ng < 4; ++ng) {
      V16U bu;
      const int base = (ng * 16 + ln) * 40 + 16 * hi;  // col, K=16*hi..+15
      bu.h[0] = *(const v8bf*)&Bs[base];
      bu.h[1] = *(const v8bf*)&Bs[base + 8];
      acc[ng] = __builtin_amdgcn_wmma_f32_16x16x32_bf16(
          false, a, false, bu.v, (short)0, acc[ng], false, false);
    }
    __syncthreads();
  }

  // ---- scatter to q / k / vT layouts
#pragma unroll
  for (int ng = 0; ng < 4; ++ng) {
#pragma unroll
    for (int r = 0; r < 8; ++r) {
      const int m = blockIdx.y * 128 + wave * 16 + r + 8 * hi;
      const int c = cb + ng * 16 + ln;
      const int b = m >> 10, i = m & 1023;
      const float val = acc[ng][r];
      if (c < 768) {
        const int h = c >> 6, d = c & 63;
        q[(((size_t)(b * NHEADS + h)) * SEQ + i) * DHEAD + d] = tobf(val * QKSCALE);
      } else if (c < 1536) {
        const int cc = c - 768, h = cc >> 6, d = cc & 63;
        k[(((size_t)(b * NHEADS + h)) * SEQ + i) * DHEAD + d] = tobf(val);
      } else {
        const int cc = c - 1536, h = cc >> 6, d = cc & 63;
        vT[(((size_t)(b * NHEADS + h)) * DHEAD + d) * SEQ + i] = tobf(val);  // [b,h,d,n]
      }
    }
  }
}

// ---------------------------------------------------------------------------
// K2: two-pass flash attention with talking heads.
// Block = (batch b, 32-query tile), 384 thr = 12 waves; wave g owns head g.
// Pass 1: exact row max / exp-sum of pre-mixed scores (S exchanged via LDS).
// Pass 2: recompute S, normalize, post-mix across heads via shared P plane,
//         accumulate O with WMMA. V tiles arrive via async DMA to LDS
//         (ASYNCcnt), issued at iteration start and consumed after the
//         4-barrier mixing phase -- latency fully hidden.
// ---------------------------------------------------------------------------
__global__ __launch_bounds__(384) void talking_attn(
    const bf16* __restrict__ q, const bf16* __restrict__ k,
    const bf16* __restrict__ vT, const float* __restrict__ mix_pre,
    const float* __restrict__ mix_post, bf16* __restrict__ ctx) {
  __shared__ float Ssh[32 * 32 * NHEADS];     // scores [i][j][h], 48 KB
  __shared__ bf16 Abuf[NHEADS][32 * 34];      // per-wave mixed-P staging
  __shared__ bf16 Vbuf[NHEADS][64 * 32];      // per-wave async V tile [d][key]
  bf16* Ps = (bf16*)Ssh;                      // P overlay [i][j][16], 32 KB

  const int tid = threadIdx.x, lane = tid & 31, g = tid >> 5;  // wave == head
  const int hi = lane >> 4, ln = lane & 15;
  const int b = blockIdx.y, i0 = blockIdx.x * 32;

  float mpre[NHEADS], mpost[NHEADS];
#pragma unroll
  for (int h = 0; h < NHEADS; ++h) {
    mpre[h] = mix_pre[h * NHEADS + g];    // column g
    mpost[h] = mix_post[h * NHEADS + g];  // column g
  }

  // Q A-fragments resident in VGPRs: rows i0..i0+31, K = d split in 2 chunks.
  v16bf aq[2][2];
#pragma unroll
  for (int mg = 0; mg < 2; ++mg)
#pragma unroll
    for (int kc = 0; kc < 2; ++kc)
#pragma unroll
      for (int v = 0; v < 8; ++v) {
        const bf16* p = &q[(((size_t)(b * NHEADS + g)) * SEQ + i0 + mg * 16 + ln) * DHEAD +
                           kc * 32 + a_kbase(v, hi)];
        v2bf t = *(const v2bf*)p;
        aq[mg][kc][2 * v] = t[0];
        aq[mg][kc][2 * v + 1] = t[1];
      }

  float m[2][8], l[2][8];
#pragma unroll
  for (int mg = 0; mg < 2; ++mg)
#pragma unroll
    for (int r = 0; r < 8; ++r) { m[mg][r] = -3.0e38f; l[mg][r] = 0.f; }

  // Async DMA constants for this wave's V tile (64 rows x 64B, stride SEQ*2B).
  const unsigned vbufLds = (unsigned)(uintptr_t)&Vbuf[g][0];
  const unsigned long long vTsaddr = (unsigned long long)(uintptr_t)vT;
  const size_t vGBase = ((size_t)(b * NHEADS + g)) * DHEAD * SEQ;  // elem index
  const int acg = lane & 3;   // 16B column group within a 64B row
  const int ar0 = lane >> 2;  // row-in-octet

  // S = Q_g @ K_g^T for a 32-key tile (B-fragments are contiguous k rows).
  auto computeS = [&](int j0, v8f s[2][2]) {
#pragma unroll
    for (int mg = 0; mg < 2; ++mg)
#pragma unroll
      for (int ng = 0; ng < 2; ++ng) s[mg][ng] = v8f_zero();
#pragma unroll
    for (int kc = 0; kc < 2; ++kc) {
      v16bf kb[2];
#pragma unroll
      for (int ng = 0; ng < 2; ++ng)
        kb[ng] = *(const v16bf*)&k[(((size_t)(b * NHEADS + g)) * SEQ + j0 + ng * 16 + ln) *
                                       DHEAD + kc * 32 + 16 * hi];
#pragma unroll
      for (int mg = 0; mg < 2; ++mg)
#pragma unroll
        for (int ng = 0; ng < 2; ++ng)
          s[mg][ng] = __builtin_amdgcn_wmma_f32_16x16x32_bf16(
              false, aq[mg][kc], false, kb[ng], (short)0, s[mg][ng], false, false);
    }
  };
  // publish this wave's S plane into Ssh[i][j][h]
  auto publishS = [&](v8f s[2][2]) {
#pragma unroll
    for (int mg = 0; mg < 2; ++mg)
#pragma unroll
      for (int ng = 0; ng < 2; ++ng)
#pragma unroll
        for (int r = 0; r < 8; ++r)
          Ssh[((mg * 16 + r + 8 * hi) * 32 + ng * 16 + ln) * NHEADS + g] = s[mg][ng][r];
  };
  // pre-softmax mix at this lane's (i,j) position: sum_h mpre[h]*S[h]
  auto mixPre = [&](int mg, int ng, int r) -> float {
    const float* sp = &Ssh[((mg * 16 + r + 8 * hi) * 32 + ng * 16 + ln) * NHEADS];
    f32x4 s0 = *(const f32x4*)sp;
    f32x4 s1 = *(const f32x4*)(sp + 4);
    f32x4 s2 = *(const f32x4*)(sp + 8);
    float a = s0[0] * mpre[0] + s0[1] * mpre[1] + s0[2] * mpre[2] + s0[3] * mpre[3];
    a += s1[0] * mpre[4] + s1[1] * mpre[5] + s1[2] * mpre[6] + s1[3] * mpre[7];
    a += s2[0] * mpre[8] + s2[1] * mpre[9] + s2[2] * mpre[10] + s2[3] * mpre[11];
    return a;
  };

  // ======================= PASS 1: softmax statistics =======================
  for (int j0 = 0; j0 < SEQ; j0 += 32) {
    v8f s[2][2];
    computeS(j0, s);
    publishS(s);
    __syncthreads();
#pragma unroll
    for (int mg = 0; mg < 2; ++mg) {
      float val[2][8];
#pragma unroll
      for (int ng = 0; ng < 2; ++ng)
#pragma unroll
        for (int r = 0; r < 8; ++r) val[ng][r] = mixPre(mg, ng, r);
#pragma unroll
      for (int r = 0; r < 8; ++r) {
        float rm = fmaxf(val[0][r], val[1][r]);
#pragma unroll
        for (int mask = 8; mask >= 1; mask >>= 1) rm = fmaxf(rm, __shfl_xor(rm, mask, 32));
        const float mn = fmaxf(m[mg][r], rm);
        float ps = __expf(val[0][r] - mn) + __expf(val[1][r] - mn);
#pragma unroll
        for (int mask = 8; mask >= 1; mask >>= 1) ps += __shfl_xor(ps, mask, 32);
        l[mg][r] = l[mg][r] * __expf(m[mg][r] - mn) + ps;
        m[mg][r] = mn;
      }
    }
    __syncthreads();  // before next tile overwrites Ssh
  }

  float linv[2][8];
#pragma unroll
  for (int mg = 0; mg < 2; ++mg)
#pragma unroll
    for (int r = 0; r < 8; ++r) linv[mg][r] = 1.f / l[mg][r];

  // ======================= PASS 2: P-mixing + PV ===========================
  v8f O[2][4];
#pragma unroll
  for (int mg = 0; mg < 2; ++mg)
#pragma unroll
    for (int ng = 0; ng < 4; ++ng) O[mg][ng] = v8f_zero();

  for (int j0 = 0; j0 < SEQ; j0 += 32) {
    // --- kick off async DMA of this wave's V tile (64x32 bf16, 8 x 512B) ---
    // Consumed only after the softmax/mixing barriers below: latency hidden.
#pragma unroll
    for (int t = 0; t < 8; ++t) {
      const int d = t * 8 + ar0;
      const unsigned ldsAddr = vbufLds + (unsigned)(d * 64 + acg * 16);
      const unsigned gOff = (unsigned)((vGBase + (size_t)d * SEQ + j0) * 2) + acg * 16;
      asm volatile("global_load_async_to_lds_b128 %0, %1, %2"
                   :: "v"(ldsAddr), "v"(gOff), "s"(vTsaddr)
                   : "memory");
    }

    v8f s[2][2];
    computeS(j0, s);
    publishS(s);
    __syncthreads();  // (1) S visible

    float pv[2][2][8];
#pragma unroll
    for (int mg = 0; mg < 2; ++mg)
#pragma unroll
      for (int ng = 0; ng < 2; ++ng)
#pragma unroll
        for (int r = 0; r < 8; ++r)
          pv[mg][ng][r] = __expf(mixPre(mg, ng, r) - m[mg][r]) * linv[mg][r];
    __syncthreads();  // (2) all S reads done; Ps overlays Ssh

#pragma unroll
    for (int mg = 0; mg < 2; ++mg)
#pragma unroll
      for (int ng = 0; ng < 2; ++ng)
#pragma unroll
        for (int r = 0; r < 8; ++r)
          Ps[((mg * 16 + r + 8 * hi) * 32 + ng * 16 + ln) * 16 + g] = tobf(pv[mg][ng][r]);
    __syncthreads();  // (3) P visible

    // post-softmax mix: a_g = sum_h mpost[h] * P[h]; stage to wave-private Abuf
#pragma unroll
    for (int mg = 0; mg < 2; ++mg)
#pragma unroll
      for (int ng = 0; ng < 2; ++ng)
#pragma unroll
        for (int r = 0; r < 8; ++r) {
          const bf16* pp = &Ps[((mg * 16 + r + 8 * hi) * 32 + ng * 16 + ln) * 16];
          union { v8bf v; bf16 e[8]; } p0;
          union { v4bf v; bf16 e[4]; } p1;
          p0.v = *(const v8bf*)pp;
          p1.v = *(const v4bf*)(pp + 8);
          float a = 0.f;
#pragma unroll
          for (int h = 0; h < 8; ++h) a += mpost[h] * (float)p0.e[h];
#pragma unroll
          for (int h = 0; h < 4; ++h) a += mpost[8 + h] * (float)p1.e[h];
          Abuf[g][(mg * 16 + r + 8 * hi) * 34 + ng * 16 + ln] = tobf(a);
        }
    __syncthreads();  // (4) all P reads done before next tile's S overwrites

    // O += A_g @ V_g  (A from Abuf, B from async-staged Vbuf[d][key])
    v16bf af[2];
#pragma unroll
    for (int mg = 0; mg < 2; ++mg)
#pragma unroll
      for (int v = 0; v < 8; ++v) {
        v2bf t = *(const v2bf*)&Abuf[g][(mg * 16 + ln) * 34 + a_kbase(v, hi)];
        af[mg][2 * v] = t[0];
        af[mg][2 * v + 1] = t[1];
      }
    wait_async0();  // V tile landed in LDS
#pragma unroll
    for (int ng = 0; ng < 4; ++ng) {
      V16U vu;
      const int vbase = (ng * 16 + ln) * 32 + 16 * hi;  // row d, keys 16*hi..+15
      vu.h[0] = *(const v8bf*)&Vbuf[g][vbase];
      vu.h[1] = *(const v8bf*)&Vbuf[g][vbase + 8];
#pragma unroll
      for (int mg = 0; mg < 2; ++mg)
        O[mg][ng] = __builtin_amdgcn_wmma_f32_16x16x32_bf16(
            false, af[mg], false, vu.v, (short)0, O[mg][ng], false, false);
    }
  }

  // ctx layout: [b, n, h*d] bf16, ready for the output projection GEMM.
#pragma unroll
  for (int mg = 0; mg < 2; ++mg)
#pragma unroll
    for (int ng = 0; ng < 4; ++ng)
#pragma unroll
      for (int r = 0; r < 8; ++r) {
        const int i = i0 + mg * 16 + r + 8 * hi;
        const int d = ng * 16 + ln;
        ctx[((size_t)(b * SEQ + i)) * INNER + g * DHEAD + d] = tobf(O[mg][ng][r]);
      }
}

// ---------------------------------------------------------------------------
// K3: ctx(bf16) [4096 x 768] @ Wout [768 x 768] + bout -> out f32.
// Same tiling as K1 (A bf16 direct from global, B f32->bf16 staged in LDS).
// ---------------------------------------------------------------------------
__global__ __launch_bounds__(256) void out_gemm(
    const bf16* __restrict__ ctx, const float* __restrict__ Wout,
    const float* __restrict__ bout, float* __restrict__ out) {
  __shared__ bf16 Bs[64 * 40];

  const int tid = threadIdx.x;
  const int lane = tid & 31, wave = tid >> 5;
  const int hi = (lane >> 4) & 1, ln = lane & 15;
  const int cb = blockIdx.x * 64;
  const int mrow = blockIdx.y * 128 + wave * 16 + ln;
  const int scol = tid & 63, soct = tid >> 6;

  v8f acc[4];
#pragma unroll
  for (int ng = 0; ng < 4; ++ng) acc[ng] = v8f_zero();

  for (int ks = 0; ks < INNER; ks += 32) {
    {
      bf16 tmp[8];
      const float* wp = &Wout[(size_t)(ks + soct * 8) * DIM + cb + scol];
#pragma unroll
      for (int j = 0; j < 8; ++j) tmp[j] = tobf(wp[(size_t)j * DIM]);
      *(v8bf*)&Bs[scol * 40 + soct * 8] = *(const v8bf*)tmp;
      if (ks + 32 < INNER)
        __builtin_prefetch(&Wout[(size_t)(ks + 32 + soct * 8) * DIM + cb + scol], 0, 1);
    }
    __syncthreads();

    v16bf a;
#pragma unroll
    for (int v = 0; v < 8; ++v) {
      v2bf p = *(const v2bf*)&ctx[(size_t)mrow * INNER + ks + a_kbase(v, hi)];
      a[2 * v] = p[0];
      a[2 * v + 1] = p[1];
    }
#pragma unroll
    for (int ng = 0; ng < 4; ++ng) {
      V16U bu;
      const int base = (ng * 16 + ln) * 40 + 16 * hi;
      bu.h[0] = *(const v8bf*)&Bs[base];
      bu.h[1] = *(const v8bf*)&Bs[base + 8];
      acc[ng] = __builtin_amdgcn_wmma_f32_16x16x32_bf16(
          false, a, false, bu.v, (short)0, acc[ng], false, false);
    }
    __syncthreads();
  }

#pragma unroll
  for (int ng = 0; ng < 4; ++ng)
#pragma unroll
    for (int r = 0; r < 8; ++r) {
      const int mo = blockIdx.y * 128 + wave * 16 + r + 8 * hi;
      const int c = cb + ng * 16 + ln;
      out[(size_t)mo * DIM + c] = acc[ng][r] + bout[c];
    }
}

// ---------------------------------------------------------------------------
extern "C" void kernel_launch(void* const* d_in, const int* in_sizes, int n_in,
                              void* d_out, int out_size, void* d_ws, size_t ws_size,
                              hipStream_t stream) {
  const float* x     = (const float*)d_in[0];
  const float* Wq    = (const float*)d_in[1];
  const float* Wkv   = (const float*)d_in[2];
  const float* mpre  = (const float*)d_in[3];
  const float* mpost = (const float*)d_in[4];
  const float* Wout  = (const float*)d_in[5];
  const float* bout  = (const float*)d_in[6];
  float* out = (float*)d_out;

  char* ws = (char*)d_ws;
  const size_t qBytes = (size_t)NBATCH * NHEADS * SEQ * DHEAD * sizeof(bf16);  // 6 MB
  bf16* q   = (bf16*)(ws);
  bf16* kk  = (bf16*)(ws + qBytes);
  bf16* vT  = (bf16*)(ws + 2 * qBytes);
  bf16* ctx = (bf16*)(ws + 3 * qBytes);

  // K1: M=4096, N=2304 -> grid (36, 32)
  qkv_gemm<<<dim3(36, 32), dim3(256), 0, stream>>>(x, Wq, Wkv, q, kk, vT);
  // K2: 32 query-tiles x 4 batches, 12 waves (heads) per block
  talking_attn<<<dim3(32, 4), dim3(384), 0, stream>>>(q, kk, vT, mpre, mpost, ctx);
  // K3: M=4096, N=768 -> grid (12, 32)
  out_gemm<<<dim3(12, 32), dim3(256), 0, stream>>>(ctx, Wout, bout, out);
}
